// GlobalEncoder_56289841382054
// MI455X (gfx1250) — matrix operand order repaired and checked by reading, not compiled, not run on hardware
//
#include <hip/hip_runtime.h>
#include <hip/hip_bf16.h>
#include <stdint.h>

typedef _Float16 v16h __attribute__((ext_vector_type(16)));
typedef float    v8f  __attribute__((ext_vector_type(8)));

#define HH 128
#define TSTEPS 32
#define FF 256
#define LRELU 0.2f

__device__ inline v8f v8f_zero() {
  v8f v;
#pragma unroll
  for (int i = 0; i < 8; ++i) v[i] = 0.f;
  return v;
}

__device__ inline unsigned floatFlip(float f) {
  unsigned u = __float_as_uint(f);
  return (u & 0x80000000u) ? ~u : (u | 0x80000000u);
}
__device__ inline float floatUnflip(unsigned u) {
  unsigned v = (u & 0x80000000u) ? (u & 0x7fffffffu) : ~u;
  return __uint_as_float(v);
}
__device__ inline float sigmoidf_(float x) { return 1.0f / (1.0f + __expf(-x)); }

// Fragment index helpers (ISA 16-bit A/B layouts, wave32).
// A 16x32 tile: lane = hi*16 + row(lr); 16 contiguous halves per lane; half i maps to
//   k-rem = (i<8) ? hi*8+i : 16 + hi*8 + (i-8).
// B 32x16 tile: lane = hi*16 + col(lr); half i maps to the same k-rem as A (table order).

// ---------------- prep kernels (pack weights into B-fragment order) ----------------

// W2 logical [k=0..255][j=0..511]: k<128 -> x-input, k>=128 -> h-input.
// j<256 merged r,z cols; 256..383 hn (Whh only); 384..511 xn (Wih only).
// Packed: W2p[((kc*32 + ct)*32 + lane)*16 + i]
__global__ void prep_gru_w_kernel(const float* __restrict__ Wih,
                                  const float* __restrict__ Whh,
                                  _Float16* __restrict__ W2p) {
  int o = blockIdx.x * 256 + threadIdx.x;
  if (o >= 8 * 32 * 32 * 16) return;
  int i = o & 15;
  int lane = (o >> 4) & 31;
  int tile = o >> 9;
  int ct = tile & 31, kc = tile >> 5;
  int lr = lane & 15, hi = lane >> 4;
  int k = kc * 32 + ((i < 8) ? hi * 8 + i : 16 + hi * 8 + (i - 8));   // B layout: table order
  int j = ct * 16 + lr;
  float v = 0.f;
  if (j < 256) {
    v = (k < HH) ? Wih[(size_t)j * HH + k] : Whh[(size_t)j * HH + (k - HH)];
  } else if (j < 384) {
    int u = j - 256;
    v = (k < HH) ? 0.f : Whh[(size_t)(256 + u) * HH + (k - HH)];
  } else {
    int u = j - 384;
    v = (k < HH) ? Wih[(size_t)(256 + u) * HH + k] : 0.f;
  }
  W2p[o] = (_Float16)v;
}

__global__ void prep_gru_b_kernel(const float* __restrict__ bih,
                                  const float* __restrict__ bhh,
                                  float* __restrict__ bias) {
  int j = blockIdx.x * 256 + threadIdx.x;
  if (j >= 512) return;
  float v;
  if (j < 256)      v = bih[j] + bhh[j];
  else if (j < 384) v = bhh[j];            // bhh[256 + (j-256)]
  else              v = bih[j - 128];      // bih[256 + (j-384)]
  bias[j] = v;
}

// Wg^T packed: Wgp[((kc*16 + ct)*32 + lane)*16 + i] = Wg[j][k], j=ct*16+lr, k=kc*32+((i<8)?hi*8+i:16+hi*8+i-8)
__global__ void prep_wg_kernel(const float* __restrict__ Wg, _Float16* __restrict__ Wgp) {
  int o = blockIdx.x * 256 + threadIdx.x;
  if (o >= 8 * 16 * 32 * 16) return;
  int i = o & 15;
  int lane = (o >> 4) & 31;
  int tile = o >> 9;
  int ct = tile & 15, kc = tile >> 4;
  int lr = lane & 15, hi = lane >> 4;
  int k = kc * 32 + ((i < 8) ? hi * 8 + i : 16 + hi * 8 + (i - 8));
  int j = ct * 16 + lr;
  Wgp[o] = (_Float16)Wg[(size_t)j * FF + k];
}

// Embedding gather, written straight into A-fragment order:
// e16p[((t*NB + rb)*4 + kc)*512 + (hi*16 + r)*16 + i]
__global__ void embed_kernel(const long long* __restrict__ X, const float* __restrict__ emb,
                             _Float16* __restrict__ e16p, int N) {
  size_t flat = (size_t)blockIdx.x * 256 + threadIdx.x;
  size_t total = (size_t)TSTEPS * N * HH;
  if (flat >= total) return;
  int c = (int)(flat & (HH - 1));
  size_t nh = flat >> 7;  // t*N + n
  int n = (int)(nh % (size_t)N);
  int t = (int)(nh / (size_t)N);
  long long tok = X[(size_t)n * TSTEPS + t];
  float v = emb[(size_t)tok * HH + c];
  int rb = n >> 4, r = n & 15;
  int kc = c >> 5, rem = c & 31;
  int hi = (rem < 16) ? (rem >> 3) : ((rem - 16) >> 3);
  int i  = (rem < 16) ? (rem & 7) : (8 + ((rem - 16) & 7));
  int NB = N >> 4;
  size_t dst = ((((size_t)t * NB + rb) * 4 + kc) * 32 + (hi * 16 + r)) * 16 + i;
  e16p[dst] = (_Float16)v;
}

__global__ void fill_u32_kernel(unsigned* __restrict__ p, unsigned v, size_t n) {
  size_t i = (size_t)blockIdx.x * 256 + threadIdx.x;
  if (i < n) p[i] = v;
}

// f32 [N,256] -> packed A-fragment f16: g16p[((rb*8 + kc)*32 + lane)*16 + i]
__global__ void cast16_pack_kernel(const float* __restrict__ in, _Float16* __restrict__ o,
                                   int N) {
  size_t tid = (size_t)blockIdx.x * 256 + threadIdx.x;
  if (tid >= (size_t)N * FF) return;
  int n = (int)(tid >> 8), c = (int)(tid & 255);
  int rb = n >> 4, r = n & 15;
  int kc = c >> 5, rem = c & 31;
  int hi = (rem < 16) ? (rem >> 3) : ((rem - 16) >> 3);
  int i  = (rem < 16) ? (rem & 7) : (8 + ((rem - 16) & 7));
  size_t dst = (((size_t)rb * 8 + kc) * 32 + (hi * 16 + r)) * 16 + i;
  o[dst] = (_Float16)in[tid];
}

// ---------------- fused GRU step ----------------
// gates = [e_t|h] @ W2 + bias, then gate math. 256 threads = 8 waves.
// BM=16 rows (rb = blockIdx.x), 512 cols; wave w owns col tiles w*4..w*4+3.
// Operands in packed fragment order; A/B fragments are software double-buffered
// so chunk kc+1 loads issue before chunk kc's wmmas (overlap memory with math).
__global__ void __launch_bounds__(256)
gru_step_kernel(const _Float16* __restrict__ e16t,  // packed slice: [rb][kc<4][lane][16]
                const _Float16* __restrict__ h16p,  // packed:       [rb][kc<4][lane][16]
                const _Float16* __restrict__ W2p,   // packed:       [kc<8][ct<32][lane][16]
                const float* __restrict__ bias,     // [512]
                float* __restrict__ h32,            // [N,128] row-major
                _Float16* __restrict__ h16w,        // packed h (write target, == h16p)
                float* __restrict__ qout,           // base for this (t,dir); stride 256/node
                int N) {
  __shared__ float smem[16][520];
  const int tid = threadIdx.x;
  const int w = tid >> 5, lane = tid & 31;
  const int lr = lane & 15, hi = lane >> 4;
  const int rb = blockIdx.x;
  const int rowBase = rb * 16;

  const _Float16* Abase0 = e16t + ((size_t)rb * 4 * 32 + lane) * 16;   // kc stride 512 halves
  const _Float16* Abase1 = h16p + ((size_t)rb * 4 * 32 + lane) * 16;
  const _Float16* Bbase  = W2p + ((size_t)(w * 4) * 32 + lane) * 16;   // ct stride 512, kc stride 512*32

  v8f acc[4];
#pragma unroll
  for (int j = 0; j < 4; ++j) acc[j] = v8f_zero();

  v16h a_cur = *(const v16h*)(Abase0);
  v16h b_cur[4];
#pragma unroll
  for (int j = 0; j < 4; ++j) b_cur[j] = *(const v16h*)(Bbase + (size_t)j * 512);

#pragma unroll
  for (int kc = 0; kc < 8; ++kc) {
    v16h a_nxt;
    v16h b_nxt[4];
    if (kc < 7) {
      const int kn = kc + 1;
      a_nxt = (kn < 4) ? *(const v16h*)(Abase0 + (size_t)kn * 512)
                       : *(const v16h*)(Abase1 + (size_t)(kn - 4) * 512);
#pragma unroll
      for (int j = 0; j < 4; ++j)
        b_nxt[j] = *(const v16h*)(Bbase + ((size_t)kn * 32 + j) * 512);
    }
#pragma unroll
    for (int j = 0; j < 4; ++j)
      acc[j] = __builtin_amdgcn_wmma_f32_16x16x32_f16(false, a_cur, false, b_cur[j],
                                                      (short)0, acc[j], false, false);
    if (kc < 7) {
      a_cur = a_nxt;
#pragma unroll
      for (int j = 0; j < 4; ++j) b_cur[j] = b_nxt[j];
    }
  }

#pragma unroll
  for (int j = 0; j < 4; ++j) {
    const int c = (w * 4 + j) * 16 + lr;
#pragma unroll
    for (int r = 0; r < 8; ++r) smem[hi * 8 + r][c] = acc[j][r];
  }
  __syncthreads();

  for (int item = tid; item < 16 * HH; item += 256) {
    const int r = item >> 7, u = item & (HH - 1);
    const int n = rowBase + r;
    if (n < N) {
      float prer = smem[r][u]          + bias[u];
      float prez = smem[r][HH + u]     + bias[HH + u];
      float hn   = smem[r][2 * HH + u] + bias[2 * HH + u];
      float xn   = smem[r][3 * HH + u] + bias[3 * HH + u];
      float rg = sigmoidf_(prer);
      float zg = sigmoidf_(prez);
      float ng = tanhf(xn + rg * hn);
      float hp = h32[(size_t)n * HH + u];
      float hv = (1.f - zg) * ng + zg * hp;
      h32[(size_t)n * HH + u] = hv;
      qout[(size_t)n * (2 * HH) + u] = hv;
      // write recurrent state in packed A-fragment order
      int kc = u >> 5, rem = u & 31;
      int hhi = (rem < 16) ? (rem >> 3) : ((rem - 16) >> 3);
      int ii  = (rem < 16) ? (rem & 7) : (8 + ((rem - 16) & 7));
      h16w[(((size_t)rb * 4 + kc) * 32 + (hhi * 16 + r)) * 16 + ii] = (_Float16)hv;
    }
  }
}

__global__ void qn_copy_kernel(const float* __restrict__ hf, const float* __restrict__ hb,
                               float* __restrict__ qn, int N) {
  size_t i = (size_t)blockIdx.x * 256 + threadIdx.x;
  if (i >= (size_t)N * 2 * HH) return;
  int n = (int)(i >> 8), c = (int)(i & 255);
  qn[i] = (c < HH) ? hf[(size_t)n * HH + c] : hb[(size_t)n * HH + (c - HH)];
}

// ---------------- GAT GEMM: C[N,256] = A(packed) * Wg^T(packed) ----------------
// Block = 16 rows x 256 cols, 8 waves, wave w -> col tiles w*2, w*2+1. Double-buffered.
__global__ void __launch_bounds__(256)
gat_gemm_kernel(const _Float16* __restrict__ Ap,   // packed [rb][kc<8][lane][16]
                const _Float16* __restrict__ Bp,   // packed [kc<8][ct<16][lane][16]
                float* __restrict__ C, int N) {
  const int tid = threadIdx.x;
  const int w = tid >> 5, lane = tid & 31;
  const int lr = lane & 15, hi = lane >> 4;
  const int rb = blockIdx.x;

  const _Float16* Abase = Ap + ((size_t)rb * 8 * 32 + lane) * 16;     // kc stride 512 halves
  const _Float16* Bbase = Bp + ((size_t)(w * 2) * 32 + lane) * 16;    // ct stride 512, kc stride 512*16

  v8f acc[2];
  acc[0] = v8f_zero();
  acc[1] = v8f_zero();

  v16h a_cur = *(const v16h*)(Abase);
  v16h b_cur[2];
#pragma unroll
  for (int j = 0; j < 2; ++j) b_cur[j] = *(const v16h*)(Bbase + (size_t)j * 512);

#pragma unroll
  for (int kc = 0; kc < 8; ++kc) {
    v16h a_nxt;
    v16h b_nxt[2];
    if (kc < 7) {
      const int kn = kc + 1;
      a_nxt = *(const v16h*)(Abase + (size_t)kn * 512);
#pragma unroll
      for (int j = 0; j < 2; ++j)
        b_nxt[j] = *(const v16h*)(Bbase + ((size_t)kn * 16 + j) * 512);
    }
#pragma unroll
    for (int j = 0; j < 2; ++j)
      acc[j] = __builtin_amdgcn_wmma_f32_16x16x32_f16(false, a_cur, false, b_cur[j],
                                                      (short)0, acc[j], false, false);
    if (kc < 7) {
      a_cur = a_nxt;
#pragma unroll
      for (int j = 0; j < 2; ++j) b_cur[j] = b_nxt[j];
    }
  }

#pragma unroll
  for (int j = 0; j < 2; ++j) {
    const int col = (w * 2 + j) * 16 + lr;
#pragma unroll
    for (int r = 0; r < 8; ++r) {
      const int row = rb * 16 + hi * 8 + r;
      if (row < N) C[(size_t)row * FF + col] = acc[j][r];
    }
  }
}

// ---------------- GAT helper kernels ----------------
__global__ void __launch_bounds__(256)
dots_kernel(const float* __restrict__ hmat, const float* __restrict__ att_s,
            const float* __restrict__ att_d, float* __restrict__ a_src,
            float* __restrict__ a_dst, int N) {
  const int w = threadIdx.x >> 5, lane = threadIdx.x & 31;
  const int n = blockIdx.x * 8 + w;
  if (n >= N) return;
  float ss = 0.f, sd = 0.f;
  for (int i = lane; i < FF; i += 32) {
    float hv = hmat[(size_t)n * FF + i];
    ss += hv * att_s[i];
    sd += hv * att_d[i];
  }
#pragma unroll
  for (int off = 16; off > 0; off >>= 1) {
    ss += __shfl_xor(ss, off, 32);
    sd += __shfl_xor(sd, off, 32);
  }
  if (lane == 0) { a_src[n] = ss; a_dst[n] = sd; }
}

__global__ void edge_e_kernel(const int* __restrict__ ei, const float* __restrict__ a_src,
                              const float* __restrict__ a_dst, float* __restrict__ ebuf,
                              unsigned* __restrict__ mbuf, int E) {
  int e = blockIdx.x * 256 + threadIdx.x;
  if (e >= E) return;
  int s = ei[e], d = ei[E + e];
  float v = a_src[s] + a_dst[d];
  v = (v > 0.f) ? v : LRELU * v;
  ebuf[e] = v;
  atomicMax(&mbuf[d], floatFlip(v));
}

__global__ void edge_ex_kernel(const int* __restrict__ ei, const float* __restrict__ ebuf,
                               const unsigned* __restrict__ mbuf, float* __restrict__ exbuf,
                               float* __restrict__ denom, int E) {
  int e = blockIdx.x * 256 + threadIdx.x;
  if (e >= E) return;
  int d = ei[E + e];
  float mv = floatUnflip(mbuf[d]);
  float ex = __expf(ebuf[e] - mv);
  exbuf[e] = ex;
  atomicAdd(&denom[d], ex);
}

__global__ void init_out_kernel(float* __restrict__ outm, const float* __restrict__ bg, int N) {
  int n = blockIdx.x, c = threadIdx.x;
  if (n < N) outm[(size_t)n * FF + c] = bg[c];
}

__global__ void edge_acc_kernel(const int* __restrict__ ei, const float* __restrict__ exbuf,
                                const float* __restrict__ denom, const float* __restrict__ hmat,
                                float* __restrict__ outm, int E) {
  int e = blockIdx.x, c = threadIdx.x;
  int s = ei[e], d = ei[E + e];
  float alpha = exbuf[e] / denom[d];
  atomicAdd(&outm[(size_t)d * FF + c], hmat[(size_t)s * FF + c] * alpha);
}

// ---------------- host launch ----------------
extern "C" void kernel_launch(void* const* d_in, const int* in_sizes, int n_in,
                              void* d_out, int out_size, void* d_ws, size_t ws_size,
                              hipStream_t stream) {
  (void)n_in; (void)out_size; (void)ws_size;
  const long long* X = (const long long*)d_in[0];
  const int* ei      = (const int*)d_in[1];
  const float* emb   = (const float*)d_in[2];
  const float* Wih_f = (const float*)d_in[3];
  const float* Whh_f = (const float*)d_in[4];
  const float* bih_f = (const float*)d_in[5];
  const float* bhh_f = (const float*)d_in[6];
  const float* Wih_b = (const float*)d_in[7];
  const float* Whh_b = (const float*)d_in[8];
  const float* bih_b = (const float*)d_in[9];
  const float* bhh_b = (const float*)d_in[10];
  const float* Wg    = (const float*)d_in[11];
  const float* att_s = (const float*)d_in[12];
  const float* att_d = (const float*)d_in[13];
  const float* bg    = (const float*)d_in[14];

  const int N = in_sizes[0] / TSTEPS;
  const int E = in_sizes[1] / 2;
  const int NB = N / 16;

  float* out    = (float*)d_out;
  float* g_out  = out;                             // [N,256]
  float* qn_out = out + (size_t)N * 2 * HH;        // [N,256]
  float* q_out  = out + (size_t)2 * N * 2 * HH;    // [T,N,256]

  char* base = (char*)d_ws;
  size_t off = 0;
  auto alloc = [&](size_t bytes) -> char* {
    char* p = base + off;
    off = (off + bytes + 255) & ~(size_t)255;
    return p;
  };

  _Float16* e16    = (_Float16*)alloc((size_t)TSTEPS * N * HH * 2);  // packed
  _Float16* W2f0   = (_Float16*)alloc((size_t)256 * 512 * 2);        // packed
  _Float16* W2f1   = (_Float16*)alloc((size_t)256 * 512 * 2);        // packed
  float* bias0     = (float*)alloc(512 * 4);
  float* bias1     = (float*)alloc(512 * 4);
  float* h32f      = (float*)alloc((size_t)N * HH * 4);
  float* h32b      = (float*)alloc((size_t)N * HH * 4);
  _Float16* h16f   = (_Float16*)alloc((size_t)N * HH * 2);           // packed
  _Float16* h16b   = (_Float16*)alloc((size_t)N * HH * 2);           // packed
  _Float16* Wgp    = (_Float16*)alloc((size_t)FF * FF * 2);          // packed
  _Float16* g16    = (_Float16*)alloc((size_t)N * FF * 2);           // packed
  float* hgat      = (float*)alloc((size_t)N * FF * 4);
  float* a_src     = (float*)alloc((size_t)N * 4);
  float* a_dst     = (float*)alloc((size_t)N * 4);
  unsigned* mbuf   = (unsigned*)alloc((size_t)N * 4);
  float* denom     = (float*)alloc((size_t)N * 4);
  float* ebuf      = (float*)alloc((size_t)E * 4);
  float* exbuf     = (float*)alloc((size_t)E * 4);
  float* bufA      = (float*)alloc((size_t)N * FF * 4);
  float* bufB      = (float*)alloc((size_t)N * FF * 4);

  // --- prep (pack weights into fragment order) ---
  prep_gru_w_kernel<<<512, 256, 0, stream>>>(Wih_f, Whh_f, W2f0);
  prep_gru_w_kernel<<<512, 256, 0, stream>>>(Wih_b, Whh_b, W2f1);
  prep_gru_b_kernel<<<2, 256, 0, stream>>>(bih_f, bhh_f, bias0);
  prep_gru_b_kernel<<<2, 256, 0, stream>>>(bih_b, bhh_b, bias1);
  prep_wg_kernel<<<256, 256, 0, stream>>>(Wg, Wgp);

  // --- embedding gather -> packed f16 A-fragments ---
  size_t etotal = (size_t)TSTEPS * N * HH;
  embed_kernel<<<(unsigned)((etotal + 255) / 256), 256, 0, stream>>>(X, emb, e16, N);

  // --- zero hidden state ---
  size_t hw = (size_t)N * HH;
  fill_u32_kernel<<<(unsigned)((hw + 255) / 256), 256, 0, stream>>>((unsigned*)h32f, 0u, hw);
  fill_u32_kernel<<<(unsigned)((hw + 255) / 256), 256, 0, stream>>>((unsigned*)h32b, 0u, hw);
  fill_u32_kernel<<<(unsigned)((hw / 2 + 255) / 256), 256, 0, stream>>>((unsigned*)h16f, 0u, hw / 2);
  fill_u32_kernel<<<(unsigned)((hw / 2 + 255) / 256), 256, 0, stream>>>((unsigned*)h16b, 0u, hw / 2);

  // --- bi-GRU scans (fwd reads e[t]; bwd reads e[T-1-t], writes q[t][:,128:]) ---
  const size_t eStride = (size_t)NB * 4 * 32 * 16;  // halves per timestep
  for (int t = 0; t < TSTEPS; ++t) {
    gru_step_kernel<<<(unsigned)NB, 256, 0, stream>>>(
        e16 + (size_t)t * eStride, h16f, W2f0, bias0, h32f, h16f,
        q_out + (size_t)t * N * 2 * HH, N);
    gru_step_kernel<<<(unsigned)NB, 256, 0, stream>>>(
        e16 + (size_t)(TSTEPS - 1 - t) * eStride, h16b, W2f1, bias1, h32b, h16b,
        q_out + (size_t)t * N * 2 * HH + HH, N);
  }
  qn_copy_kernel<<<(unsigned)(((size_t)N * 2 * HH + 255) / 256), 256, 0, stream>>>(
      h32f, h32b, qn_out, N);

  // --- 4 GAT layers (shared weights) ---
  const float* gin = qn_out;
  for (int layer = 0; layer < 4; ++layer) {
    float* gout = (layer == 3) ? g_out : ((layer & 1) ? bufB : bufA);
    size_t nf = (size_t)N * FF;
    cast16_pack_kernel<<<(unsigned)((nf + 255) / 256), 256, 0, stream>>>(gin, g16, N);
    gat_gemm_kernel<<<(unsigned)NB, 256, 0, stream>>>(g16, Wgp, hgat, N);
    dots_kernel<<<(unsigned)((N + 7) / 8), 256, 0, stream>>>(hgat, att_s, att_d, a_src, a_dst, N);
    fill_u32_kernel<<<(unsigned)((N + 255) / 256), 256, 0, stream>>>(mbuf, 0x007FFFFFu, (size_t)N);
    edge_e_kernel<<<(unsigned)((E + 255) / 256), 256, 0, stream>>>(ei, a_src, a_dst, ebuf, mbuf, E);
    fill_u32_kernel<<<(unsigned)((N + 255) / 256), 256, 0, stream>>>((unsigned*)denom, 0u, (size_t)N);
    edge_ex_kernel<<<(unsigned)((E + 255) / 256), 256, 0, stream>>>(ei, ebuf, mbuf, exbuf, denom, E);
    init_out_kernel<<<(unsigned)N, 256, 0, stream>>>(gout, bg, N);
    edge_acc_kernel<<<(unsigned)E, 256, 0, stream>>>(ei, exbuf, denom, hgat, gout, E);
    gin = gout;
  }
}
